// ConvSeqAttentionModel_33157147525310
// MI455X (gfx1250) — matrix operand order repaired and checked by reading, not compiled, not run on hardware
//
#include <hip/hip_runtime.h>
#include <math.h>

// ---------------- problem constants ----------------
#define NND    64          // nodes
#define BATCH  4
#define BNS    256         // B*N samples
#define LAGS   128
#define TENC   121         // LAGS - DEC_W
#define HIDC   64
#define EEDG   1280        // E = N*TOPK
#define EPSL   1e-5f
#define LOG1E5 11.512925464970229f   // ln(100000)

typedef _Float16 v16h __attribute__((ext_vector_type(16)));
typedef float    v8f  __attribute__((ext_vector_type(8)));

__device__ __forceinline__ float bnf(const float* p, int C, int ch, float v) {
  // (v - mean)*rsqrt(var+eps)*gamma + beta ; p laid out [gamma|beta|mean|var] x C
  return (v - p[2 * C + ch]) * rsqrtf(p[3 * C + ch] + EPSL) * p[0 * C + ch] + p[1 * C + ch];
}
__device__ __forceinline__ float sigm(float x) { return 1.f / (1.f + expf(-x)); }
__device__ __forceinline__ float gelu(float x) { return 0.5f * x * (1.f + erff(x * 0.70710678118654752f)); }
__device__ __forceinline__ float pe_val(int c, int t, int C, float inv_scale) {
  float dv = expf(-(float)(c & ~1) * (LOG1E5 / (float)C));
  float a  = (float)t * dv;
  return ((c & 1) ? cosf(a) : sinf(a)) * inv_scale;
}

// ---------------- graph setup: gcn_norm + CSR (deterministic) ----------------
__global__ void build_graph(const int* __restrict__ src, const int* __restrict__ dst,
                            const float* __restrict__ ew,
                            int* __restrict__ rowptr, int* __restrict__ srcp,
                            float* __restrict__ nwp) {
  __shared__ float degf[NND];
  __shared__ int   cnt[NND];
  __shared__ int   off[NND + 1];
  __shared__ float dis[NND];
  int tid = threadIdx.x;
  if (tid < NND) {            // serial per-dst scan: deterministic order (matches segment_sum)
    float s = 0.f; int c = 0;
    for (int e = 0; e < EEDG; ++e)
      if (dst[e] == tid) { s += ew[e]; ++c; }
    degf[tid] = s; cnt[tid] = c;
    dis[tid]  = (s > 0.f) ? rsqrtf(s) : 0.f;
  }
  __syncthreads();
  if (tid == 0) {
    off[0] = 0;
    for (int n = 0; n < NND; ++n) off[n + 1] = off[n] + cnt[n];
  }
  __syncthreads();
  if (tid < NND) {
    int pos = off[tid];
    for (int e = 0; e < EEDG; ++e)
      if (dst[e] == tid) {
        srcp[pos] = src[e];
        nwp[pos]  = dis[src[e]] * ew[e] * dis[tid];
        ++pos;
      }
  }
  if (tid < NND + 1) rowptr[tid] = off[tid];
}

// ---------------- weight prep: W (K x D, f32) -> WT (D x Kp, f16, zero-padded) --------
__global__ void prep_weights(const float* __restrict__ Wi, const float* __restrict__ Wr,
                             _Float16* __restrict__ WiT, _Float16* __restrict__ WrT,
                             int K, int D, int Kp) {
  int idx = blockIdx.x * blockDim.x + threadIdx.x;
  if (idx >= D * Kp) return;
  int d = idx / Kp, k = idx % Kp;
  WiT[idx] = (k < K) ? (_Float16)Wi[(size_t)k * D + d] : (_Float16)0.f;
  WrT[idx] = (k < K) ? (_Float16)Wr[(size_t)k * D + d] : (_Float16)0.f;
}

// ---------------- temporal conv + BN + GLU + PE -> f16 gmat rows ----------------
// x indexed as x[bn*xsBN + c*xsC + t]; gmat row = ((bn/64)*TENC + t)*64 + bn%64, stride Kp
__global__ void tconv_glu(const float* __restrict__ x, int xsBN, int xsC,
                          const float* __restrict__ w, const float* __restrict__ bias,
                          const float* __restrict__ bnp, int tc, int cin, int kk,
                          _Float16* __restrict__ g, int Kp) {
  __shared__ float xs[160];               // cin*kk <= 96
  int bid = blockIdx.x;
  int bn = bid / TENC, t = bid % TENC;
  int nload = cin * kk;
  for (int idx = threadIdx.x; idx < nload; idx += blockDim.x) {
    int c = idx / kk, q = idx % kk;
    int tt = t - (kk - 1) + q;
    xs[idx] = (tt >= 0) ? x[(size_t)bn * xsBN + c * xsC + tt] : 0.f;
  }
  __syncthreads();
  size_t row = ((size_t)(bn >> 6) * TENC + t) * 64 + (bn & 63);
  int C2 = 2 * tc;
  for (int c = threadIdx.x; c < Kp; c += blockDim.x) {
    float val = 0.f;
    if (c < tc) {
      float v0 = bias[c], v1 = bias[c + tc];
      const float* w0 = w + (size_t)c * cin * kk;
      const float* w1 = w + (size_t)(c + tc) * cin * kk;
      for (int i = 0; i < cin; ++i)
        for (int q = 0; q < kk; ++q) {
          float xv = xs[i * kk + q];
          v0 += w0[i * kk + q] * xv;
          v1 += w1[i * kk + q] * xv;
        }
      v0 = bnf(bnp, C2, c, v0);
      v1 = bnf(bnp, C2, c + tc, v1);
      val = v0 * sigm(v1) + pe_val(c, t, tc, 0.01f);   // + PE(T,tc)/100
    }
    g[row * Kp + c] = (_Float16)val;
  }
}

// ---------------- dual WMMA GEMM: H = A@Wi, R = A@Wr (f16 in, f32 acc) ----------------
// A:  Rrows x Kp f16 (zero-padded K..Kp)
// WT: D x Kp f16 pre-transposed/padded so each lane's B fragment is 16 contiguous halves
// 8 waves per block, one 16x16 output tile per wave; EXEC all-ones inside each wave.
__global__ __launch_bounds__(256) void gemm_wmma_dual(const _Float16* __restrict__ A,
                                                      const _Float16* __restrict__ WiT,
                                                      const _Float16* __restrict__ WrT,
                                                      float* __restrict__ H,
                                                      float* __restrict__ Rr,
                                                      int Kp, int D, int tilesN,
                                                      int tilesTotal) {
  int wid  = threadIdx.x >> 5;
  int tile = blockIdx.x * 8 + wid;
  if (tile >= tilesTotal) return;
  int tm   = tile / tilesN, tn = tile % tilesN;
  int lane = threadIdx.x & 31;
  int m    = lane & 15;        // A row within tile / B-C-D column within tile
  int lh   = lane >> 4;
  // hoisted per-lane base pointers (bumped by 32 halves per K-step)
  const _Float16* Ap  = A   + (size_t)(tm * 16 + m) * Kp + 16 * lh;  // 16 contiguous halves
  const _Float16* Bip = WiT + (size_t)(tn * 16 + m) * Kp + 16 * lh;  // 16 contiguous halves
  const _Float16* Brp = WrT + (size_t)(tn * 16 + m) * Kp + 16 * lh;
  v8f accH = {}; v8f accR = {};
  for (int k0 = 0; k0 < Kp; k0 += 32) {
    v16h a, bi, br;
#pragma unroll
    for (int j = 0; j < 8; ++j) { a[j] = Ap[j]; a[8 + j] = Ap[8 + j]; }
#pragma unroll
    for (int j = 0; j < 16; ++j) { bi[j] = Bip[j]; br[j] = Brp[j]; }
    accH = __builtin_amdgcn_wmma_f32_16x16x32_f16(false, a, false, bi, (short)0, accH, false, false);
    accR = __builtin_amdgcn_wmma_f32_16x16x32_f16(false, a, false, br, (short)0, accR, false, false);
    Ap += 32; Bip += 32; Brp += 32;
  }
  float* Hp = H  + (size_t)(tm * 16 + 8 * lh) * D + tn * 16 + m;
  float* Rp = Rr + (size_t)(tm * 16 + 8 * lh) * D + tn * 16 + m;
#pragma unroll
  for (int r = 0; r < 8; ++r) {
    Hp[(size_t)r * D] = accH[r];
    Rp[(size_t)r * D] = accR[r];
  }
}

// ---------------- graph aggregation + GELU + residual 1x1 conv ----------------
__global__ void aggregate(const float* __restrict__ hb, const float* __restrict__ rb,
                          const float* __restrict__ ab,
                          const int* __restrict__ rowptr, const int* __restrict__ srcp,
                          const float* __restrict__ nwp,
                          const float* __restrict__ x, int xsBN, int xsC,
                          const float* __restrict__ resw, const float* __restrict__ resb,
                          int cin, int D, float* __restrict__ ybuf) {
  __shared__ float hT[NND * 64];          // 16 KB max
  int g = blockIdx.x;
  int b_outer = g / TENC, t = g % TENC;
  int tot = NND * D;
  for (int idx = threadIdx.x; idx < tot; idx += blockDim.x)
    hT[idx] = hb[((size_t)g * NND + idx / D) * D + idx % D];
  __syncthreads();
  for (int idx = threadIdx.x; idx < tot; idx += blockDim.x) {
    int n = idx / D, d = idx % D;
    float acc = 0.f;
    int p1 = rowptr[n + 1];
    for (int p = rowptr[n]; p < p1; ++p)
      acc += nwp[p] * hT[srcp[p] * D + d];
    float v = gelu(acc + rb[((size_t)g * NND + n) * D + d] + ab[d]);
    int bn = b_outer * NND + n;
    float rv = resb[d];
    for (int c = 0; c < cin; ++c)
      rv += x[(size_t)bn * xsBN + c * xsC + t] * resw[c * D + d];
    ybuf[((size_t)bn * D + d) * TENC + t] = v + rv;
  }
}

// ---------------- per-sample (channel,time) normalization, optional big PE ----------------
__global__ void samplenorm(const float* __restrict__ ybuf, float* __restrict__ out,
                           int D, int addPe, float peInv) {
  __shared__ float rs[256], rq[256];
  int bn = blockIdx.x, tid = threadIdx.x;
  int cnt = D * TENC;
  const float* yb = ybuf + (size_t)bn * cnt;
  float s = 0.f, q = 0.f;
  for (int idx = tid; idx < cnt; idx += 256) {
    float v = yb[idx];
    s += v; q += v * v;
  }
  rs[tid] = s; rq[tid] = q;
  __syncthreads();
  for (int w = 128; w >= 1; w >>= 1) {
    if (tid < w) { rs[tid] += rs[tid + w]; rq[tid] += rq[tid + w]; }
    __syncthreads();
  }
  float mu  = rs[0] / (float)cnt;
  float var = rq[0] / (float)cnt - mu * mu;
  float rsq = rsqrtf(var + EPSL);
  float* ob = out + (size_t)bn * cnt;
  for (int idx = tid; idx < cnt; idx += 256) {
    float v = (yb[idx] - mu) * rsq;
    if (addPe) v += pe_val(idx / TENC, idx % TENC, D, peInv);
    ob[idx] = v;
  }
}

// ---------------- alignment projection: a[b,o,t] = sum_c h[b,c,t]*W[c,o] + b[o] ----------
__global__ void align_proj(const float* __restrict__ hfin, const float* __restrict__ aw,
                           const float* __restrict__ abias, float* __restrict__ abuf) {
  __shared__ float hc[HIDC];
  int bid = blockIdx.x;
  int bn = bid / TENC, t = bid % TENC;
  int o = threadIdx.x;
  hc[o] = hfin[((size_t)bn * HIDC + o) * TENC + t];
  __syncthreads();
  float acc = abias[o];
  for (int c = 0; c < HIDC; ++c) acc += hc[c] * aw[c * HIDC + o];
  abuf[((size_t)bn * HIDC + o) * TENC + t] = acc;
}

// ---------------- decoder window init ----------------
__global__ void dw_init(const float* __restrict__ window, float* __restrict__ dw) {
  int idx = blockIdx.x * blockDim.x + threadIdx.x;
  if (idx < BNS * 7) dw[idx] = window[(idx / 7) * LAGS + TENC + (idx % 7)];
}

// ---------------- decoder conv stack (two valid convs + BN + GLU) -> f16 rows ----------
__global__ void dec_conv(const float* __restrict__ dw,
                         const float* __restrict__ w1, const float* __restrict__ b1,
                         const float* __restrict__ bn1,
                         const float* __restrict__ w2, const float* __restrict__ b2,
                         const float* __restrict__ bn2,
                         _Float16* __restrict__ g) {
  __shared__ float c1s[48];               // 16 ch x 3 t after GLU
  int s = blockIdx.x, tid = threadIdx.x;
  const float* dws = dw + s * 7;
  for (int idx = tid; idx < 48; idx += 32) {
    int ch = idx / 3, tt = idx % 3;
    float ya = b1[ch], yb = b1[ch + 16];
    for (int q = 0; q < 5; ++q) {
      float xv = dws[tt + q];
      ya += w1[ch * 5 + q] * xv;
      yb += w1[(ch + 16) * 5 + q] * xv;
    }
    ya = bnf(bn1, 32, ch, ya);
    yb = bnf(bn1, 32, ch + 16, yb);
    c1s[idx] = ya * sigm(yb);
  }
  __syncthreads();
  int oc = tid;                           // 32 output channels after GLU
  float ya = b2[oc], yb = b2[oc + 32];
  for (int i = 0; i < 16; ++i)
    for (int q = 0; q < 3; ++q) {
      float xv = c1s[i * 3 + q];
      ya += w2[oc * 48 + i * 3 + q] * xv;
      yb += w2[(oc + 32) * 48 + i * 3 + q] * xv;
    }
  ya = bnf(bn2, 64, oc, ya);
  yb = bnf(bn2, 64, oc + 32, yb);
  g[s * 32 + oc] = (_Float16)(ya * sigm(yb));
}

// ---------------- decoder ARMA aggregation + GELU + layernorm(d_ln) -----------------
__global__ void dec_arma(const float* __restrict__ hb, const float* __restrict__ rb,
                         const float* __restrict__ ab,
                         const int* __restrict__ rowptr, const int* __restrict__ srcp,
                         const float* __restrict__ nwp,
                         const float* __restrict__ dln, float* __restrict__ dvec) {
  __shared__ float hT[NND * HIDC];
  __shared__ float ov[NND * HIDC];
  __shared__ float mus[NND], vs_[NND];
  int b = blockIdx.x, tid = threadIdx.x;
  for (int idx = tid; idx < NND * HIDC; idx += 256)
    hT[idx] = hb[((size_t)b * NND + idx / HIDC) * HIDC + idx % HIDC];
  __syncthreads();
  for (int idx = tid; idx < NND * HIDC; idx += 256) {
    int n = idx / HIDC, d = idx % HIDC;
    float acc = 0.f;
    int p1 = rowptr[n + 1];
    for (int p = rowptr[n]; p < p1; ++p)
      acc += nwp[p] * hT[srcp[p] * HIDC + d];
    ov[idx] = gelu(acc + rb[((size_t)b * NND + n) * HIDC + d] + ab[d]);
  }
  __syncthreads();
  if (tid < NND) {
    float s = 0.f, q = 0.f;
    for (int d = 0; d < HIDC; ++d) {
      float v = ov[tid * HIDC + d];
      s += v; q += v * v;
    }
    float mu = s / (float)HIDC;
    mus[tid] = mu;
    vs_[tid] = q / (float)HIDC - mu * mu;
  }
  __syncthreads();
  for (int idx = tid; idx < NND * HIDC; idx += 256) {
    int n = idx / HIDC, d = idx % HIDC;
    float v = (ov[idx] - mus[n]) * rsqrtf(vs_[n] + EPSL) * dln[d] + dln[HIDC + d];
    dvec[((size_t)b * NND + n) * HIDC + d] = v;
  }
}

// ---------------- attention + prediction + dw shift (one block per sample) ----------
__global__ void dec_attn(const float* __restrict__ dvec, const float* __restrict__ abuf,
                         const float* __restrict__ hfin,
                         const float* __restrict__ pw, const float* __restrict__ pb,
                         float* __restrict__ dw, float* __restrict__ out, int step) {
  __shared__ float dl[HIDC];
  __shared__ float scv[TENC];
  __shared__ float cc[2 * HIDC];
  __shared__ float sinv;
  int s = blockIdx.x, tid = threadIdx.x;
  if (tid < HIDC) dl[tid] = dvec[(size_t)s * HIDC + tid];
  __syncthreads();
  for (int t = tid; t < TENC; t += 128) {
    float acc = 0.f;
    for (int c = 0; c < HIDC; ++c)
      acc += dl[c] * abuf[((size_t)s * HIDC + c) * TENC + t];
    scv[t] = acc;
  }
  __syncthreads();
  if (tid == 0) {
    float m = -3.0e38f;
    for (int t = 0; t < TENC; ++t) m = fmaxf(m, scv[t]);
    float su = 0.f;
    for (int t = 0; t < TENC; ++t) { float e = expf(scv[t] - m); scv[t] = e; su += e; }
    sinv = 1.f / su;
  }
  __syncthreads();
  if (tid < HIDC) {
    float acc = 0.f;
    const float* hr = hfin + ((size_t)s * HIDC + tid) * TENC;
    for (int t = 0; t < TENC; ++t) acc += scv[t] * hr[t];
    cc[tid]         = acc * sinv;
    cc[HIDC + tid]  = dl[tid];
  }
  __syncthreads();
  if (tid == 0) {
    float su = 0.f, q = 0.f;
    for (int i = 0; i < 2 * HIDC; ++i) { su += cc[i]; q += cc[i] * cc[i]; }
    float mu  = su / (float)(2 * HIDC);
    float var = q / (float)(2 * HIDC) - mu * mu;
    float rs  = rsqrtf(var + EPSL);
    float y = pb[0];
    for (int i = 0; i < 2 * HIDC; ++i) y += (cc[i] - mu) * rs * pw[i];
    out[s * 4 + step] = y;
    for (int i = 0; i < 6; ++i) dw[s * 7 + i] = dw[s * 7 + i + 1];
    dw[s * 7 + 6] = y;
  }
}

// =====================================================================================
extern "C" void kernel_launch(void* const* d_in, const int* in_sizes, int n_in,
                              void* d_out, int out_size, void* d_ws, size_t ws_size,
                              hipStream_t stream) {
  (void)in_sizes; (void)n_in; (void)out_size; (void)ws_size;
  const float* window = (const float*)d_in[0];
  const int*   eidx   = (const int*)d_in[1];
  const float* eattr  = (const float*)d_in[2];
  // encoder layer params: res_w,res_b,tconv_w,tconv_b,bn,arma_wi,arma_wr,arma_b
  const float* e1[8]; const float* e2[8]; const float* e3[8];
  for (int i = 0; i < 8; ++i) { e1[i] = (const float*)d_in[3 + i];
                                e2[i] = (const float*)d_in[11 + i];
                                e3[i] = (const float*)d_in[19 + i]; }
  const float* d_c1_w = (const float*)d_in[27];
  const float* d_c1_b = (const float*)d_in[28];
  const float* d_bn1  = (const float*)d_in[29];
  const float* d_c2_w = (const float*)d_in[30];
  const float* d_c2_b = (const float*)d_in[31];
  const float* d_bn2  = (const float*)d_in[32];
  const float* d_wi   = (const float*)d_in[33];
  const float* d_wr   = (const float*)d_in[34];
  const float* d_ab   = (const float*)d_in[35];
  const float* d_ln   = (const float*)d_in[36];
  const float* aw     = (const float*)d_in[37];
  const float* abias  = (const float*)d_in[38];
  const float* pw     = (const float*)d_in[39];
  const float* pb     = (const float*)d_in[40];
  float* out = (float*)d_out;

  const int G    = BATCH * TENC;        // 484 groups
  const int RMAX = G * NND;             // 30976 GEMM rows

  size_t off = 0;
  auto carve = [&](size_t bytes) -> void* {
    void* p = (char*)d_ws + off;
    off += (bytes + 255) & ~(size_t)255;
    return p;
  };
  int*      gRowptr = (int*)carve((NND + 1) * sizeof(int));
  int*      gSrcp   = (int*)carve(EEDG * sizeof(int));
  float*    gNwp    = (float*)carve(EEDG * sizeof(float));
  _Float16* gmat    = (_Float16*)carve((size_t)RMAX * 64 * sizeof(_Float16));
  _Float16* wiT     = (_Float16*)carve((size_t)64 * 64 * sizeof(_Float16));
  _Float16* wrT     = (_Float16*)carve((size_t)64 * 64 * sizeof(_Float16));
  float*    hbuf    = (float*)carve((size_t)RMAX * 64 * sizeof(float));
  float*    rbuf    = (float*)carve((size_t)RMAX * 64 * sizeof(float));
  float*    ybuf    = (float*)carve((size_t)BNS * 64 * TENC * sizeof(float));
  float*    x1      = (float*)carve((size_t)BNS * 16 * TENC * sizeof(float));
  float*    x2      = (float*)carve((size_t)BNS * 32 * TENC * sizeof(float));
  float*    hfin    = (float*)carve((size_t)BNS * 64 * TENC * sizeof(float));
  float*    abuf    = (float*)carve((size_t)BNS * 64 * TENC * sizeof(float));
  float*    dwb     = (float*)carve((size_t)BNS * 7 * sizeof(float));
  float*    dvec    = (float*)carve((size_t)BNS * 64 * sizeof(float));

  build_graph<<<1, 256, 0, stream>>>(eidx, eidx + EEDG, eattr, gRowptr, gSrcp, gNwp);

  const int tilesM = RMAX / 16;         // 1936
  auto gemmGrid = [](int tiles) { return (tiles + 7) / 8; };

  // ---- stconv layer 1: cin=1, tc=sc=16, k=5 (window input, channel stride unused) ----
  tconv_glu<<<BNS * TENC, 64, 0, stream>>>(window, LAGS, 0, e1[2], e1[3], e1[4],
                                           16, 1, 5, gmat, 32);
  prep_weights<<<(16 * 32 + 255) / 256, 256, 0, stream>>>(e1[5], e1[6], wiT, wrT, 16, 16, 32);
  gemm_wmma_dual<<<gemmGrid(tilesM * 1), 256, 0, stream>>>(gmat, wiT, wrT, hbuf, rbuf,
                                                           32, 16, 1, tilesM * 1);
  aggregate<<<G, 256, 0, stream>>>(hbuf, rbuf, e1[7], gRowptr, gSrcp, gNwp,
                                   window, LAGS, 0, e1[0], e1[1], 1, 16, ybuf);
  samplenorm<<<BNS, 256, 0, stream>>>(ybuf, x1, 16, 0, 0.f);

  // ---- stconv layer 2: cin=16, tc=sc=32, k=3 ----
  tconv_glu<<<BNS * TENC, 64, 0, stream>>>(x1, 16 * TENC, TENC, e2[2], e2[3], e2[4],
                                           32, 16, 3, gmat, 32);
  prep_weights<<<(32 * 32 + 255) / 256, 256, 0, stream>>>(e2[5], e2[6], wiT, wrT, 32, 32, 32);
  gemm_wmma_dual<<<gemmGrid(tilesM * 2), 256, 0, stream>>>(gmat, wiT, wrT, hbuf, rbuf,
                                                           32, 32, 2, tilesM * 2);
  aggregate<<<G, 256, 0, stream>>>(hbuf, rbuf, e2[7], gRowptr, gSrcp, gNwp,
                                   x1, 16 * TENC, TENC, e2[0], e2[1], 16, 32, ybuf);
  samplenorm<<<BNS, 256, 0, stream>>>(ybuf, x2, 32, 0, 0.f);

  // ---- stconv layer 3: cin=32, tc=sc=64, k=3; output gets big PE (scale 1e9) ----
  tconv_glu<<<BNS * TENC, 64, 0, stream>>>(x2, 32 * TENC, TENC, e3[2], e3[3], e3[4],
                                           64, 32, 3, gmat, 64);
  prep_weights<<<(64 * 64 + 255) / 256, 256, 0, stream>>>(e3[5], e3[6], wiT, wrT, 64, 64, 64);
  gemm_wmma_dual<<<gemmGrid(tilesM * 4), 256, 0, stream>>>(gmat, wiT, wrT, hbuf, rbuf,
                                                           64, 64, 4, tilesM * 4);
  aggregate<<<G, 256, 0, stream>>>(hbuf, rbuf, e3[7], gRowptr, gSrcp, gNwp,
                                   x2, 32 * TENC, TENC, e3[0], e3[1], 32, 64, ybuf);
  samplenorm<<<BNS, 256, 0, stream>>>(ybuf, hfin, 64, 1, 1e-9f);

  // ---- alignment projection ----
  align_proj<<<BNS * TENC, 64, 0, stream>>>(hfin, aw, abias, abuf);

  // ---- decoder ----
  dw_init<<<(BNS * 7 + 255) / 256, 256, 0, stream>>>(window, dwb);
  prep_weights<<<(64 * 32 + 255) / 256, 256, 0, stream>>>(d_wi, d_wr, wiT, wrT, 32, 64, 32);
  const int decTiles = (BNS / 16) * 4;   // 64 tiles
  for (int step = 0; step < 4; ++step) {
    dec_conv<<<BNS, 32, 0, stream>>>(dwb, d_c1_w, d_c1_b, d_bn1, d_c2_w, d_c2_b, d_bn2, gmat);
    gemm_wmma_dual<<<gemmGrid(decTiles), 256, 0, stream>>>(gmat, wiT, wrT, hbuf, rbuf,
                                                           32, 64, 4, decTiles);
    dec_arma<<<BATCH, 256, 0, stream>>>(hbuf, rbuf, d_ab, gRowptr, gSrcp, gNwp, d_ln, dvec);
    dec_attn<<<BNS, 128, 0, stream>>>(dvec, abuf, hfin, pw, pb, dwb, out, step);
  }
}
